// SimpleBitNetModel_16956530884786
// MI455X (gfx1250) — 8-bit matrix operand order repaired and checked by reading, compiled, not run on hardware
//
#include <hip/hip_runtime.h>
#include <stdint.h>

#define HDIM 4096
#define RDIM 8
#define MTOK 8192
#define EPSQ 1e-5f

typedef __attribute__((ext_vector_type(8))) int v8i;

__global__ void zero_scalar_k(float* p) {
  if (threadIdx.x == 0) p[0] = 0.0f;
}

__global__ void absmean_reduce_k(const float* __restrict__ w,
                                 float* __restrict__ sum, int n) {
  __shared__ float sm[256];
  float acc = 0.0f;
  for (int i = blockIdx.x * blockDim.x + threadIdx.x; i < n;
       i += gridDim.x * blockDim.x)
    acc += fabsf(w[i]);
  sm[threadIdx.x] = acc;
  __syncthreads();
  for (int s = 128; s > 0; s >>= 1) {
    if ((int)threadIdx.x < s) sm[threadIdx.x] += sm[threadIdx.x + s];
    __syncthreads();
  }
  if (threadIdx.x == 0) atomicAdd(sum, sm[0]);
}

__global__ void quant_weight_k(const float* __restrict__ w,
                               const float* __restrict__ sumptr,
                               signed char* __restrict__ wq,
                               float* __restrict__ winv_out, int n,
                               float inv_n) {
  const float mean = sumptr[0] * inv_n;
  const float clipped = fmaxf(mean, EPSQ);
  const float sw = 1.0f / clipped;
  if (blockIdx.x == 0 && threadIdx.x == 0) winv_out[0] = clipped;
  for (int i = blockIdx.x * blockDim.x + threadIdx.x; i < n;
       i += gridDim.x * blockDim.x) {
    float q = rintf(w[i] * sw);
    q = fminf(fmaxf(q, -1.0f), 1.0f);
    wq[i] = (signed char)(int)q;
  }
}

__global__ void act_quant_lora_k(const float* __restrict__ x,
                                 const float* __restrict__ la,
                                 signed char* __restrict__ xq,
                                 float* __restrict__ inv_sx,
                                 float* __restrict__ t) {
  __shared__ float sm[256];
  const int row = blockIdx.x;
  const int tid = threadIdx.x;
  const float* xr = x + (size_t)row * HDIM;
  const int base = tid * 16;

  float v[16];
  float amax = 0.0f;
#pragma unroll
  for (int i = 0; i < 16; ++i) {
    v[i] = xr[base + i];
    amax = fmaxf(amax, fabsf(v[i]));
  }

  float dots[RDIM];
#pragma unroll
  for (int r = 0; r < RDIM; ++r) {
    const float* lar = la + (size_t)r * HDIM + base;
    float d = 0.0f;
#pragma unroll
    for (int i = 0; i < 16; ++i) d = fmaf(v[i], lar[i], d);
    dots[r] = d;
  }

  sm[tid] = amax;
  __syncthreads();
  for (int s = 128; s > 0; s >>= 1) {
    if (tid < s) sm[tid] = fmaxf(sm[tid], sm[tid + s]);
    __syncthreads();
  }
  const float gmax = fmaxf(sm[0], EPSQ);
  __syncthreads();

  for (int r = 0; r < RDIM; ++r) {
    sm[tid] = dots[r];
    __syncthreads();
    for (int s = 128; s > 0; s >>= 1) {
      if (tid < s) sm[tid] += sm[tid + s];
      __syncthreads();
    }
    if (tid == 0) t[(size_t)row * RDIM + r] = sm[0];
    __syncthreads();
  }

  if (tid == 0) inv_sx[row] = gmax * (1.0f / 127.0f);

  const float sc = 127.0f / gmax;
  int pk[4];
#pragma unroll
  for (int g = 0; g < 4; ++g) {
    int word = 0;
#pragma unroll
    for (int b = 0; b < 4; ++b) {
      float q = rintf(v[g * 4 + b] * sc);
      q = fminf(fmaxf(q, -128.0f), 127.0f);
      word |= ((int)q & 0xff) << (8 * b);
    }
    pk[g] = word;
  }
  int* dst = (int*)(xq + (size_t)row * HDIM + base);
#pragma unroll
  for (int g = 0; g < 4; ++g) dst[g] = pk[g];
}

__global__ __launch_bounds__(256) void bitnet_wmma_gemm_k(
    const signed char* __restrict__ xq, const float* __restrict__ inv_sx,
    const signed char* __restrict__ wq, const float* __restrict__ winv_p,
    const float* __restrict__ t, const float* __restrict__ lb,
    float* __restrict__ out, int relu) {
  const int lane = threadIdx.x & 31;
  const int wave = threadIdx.x >> 5;
  const int wm = wave >> 1;
  const int wn = wave & 1;
  const int m0 = blockIdx.y * 128 + wm * 32;
  const int n0 = blockIdx.x * 128 + wn * 64;
  const int hi = lane >> 4;
  const int lr = lane & 15;

  const signed char* ap[2];
  ap[0] = xq + (size_t)(m0 + lr) * HDIM;
  ap[1] = ap[0] + (size_t)16 * HDIM;
  const signed char* bp[4];
#pragma unroll
  for (int nt = 0; nt < 4; ++nt)
    bp[nt] = wq + (size_t)(n0 + nt * 16 + lr) * HDIM;

  auto loadA = [&](const signed char* p) -> v8i {
    const int2 u0 = *(const int2*)(p + 0 + hi * 8);
    const int2 u1 = *(const int2*)(p + 16 + hi * 8);
    const int2 u2 = *(const int2*)(p + 32 + hi * 8);
    const int2 u3 = *(const int2*)(p + 48 + hi * 8);
    v8i r = {u0.x, u0.y, u1.x, u1.y, u2.x, u2.y, u3.x, u3.y};
    return r;
  };
  auto loadB = [&](const signed char* p) -> v8i {
    const int2 q0 = *(const int2*)(p + 0 + hi * 8);
    const int2 q1 = *(const int2*)(p + 16 + hi * 8);
    const int2 q2 = *(const int2*)(p + 32 + hi * 8);
    const int2 q3 = *(const int2*)(p + 48 + hi * 8);
    v8i r = {q0.x, q0.y, q1.x, q1.y, q2.x, q2.y, q3.x, q3.y};
    return r;
  };

  v8i acc[2][4];
#pragma unroll
  for (int mt = 0; mt < 2; ++mt)
#pragma unroll
    for (int nt = 0; nt < 4; ++nt) acc[mt][nt] = (v8i){};

  for (int k = 0; k < HDIM; k += 64) {
    v8i a[2], b[4];
#pragma unroll
    for (int mt = 0; mt < 2; ++mt) a[mt] = loadA(ap[mt] + k);
#pragma unroll
    for (int nt = 0; nt < 4; ++nt) b[nt] = loadB(bp[nt] + k);
#pragma unroll
    for (int mt = 0; mt < 2; ++mt)
#pragma unroll
      for (int nt = 0; nt < 4; ++nt)
        acc[mt][nt] = __builtin_amdgcn_wmma_i32_16x16x64_iu8(
            true, a[mt], true, b[nt], acc[mt][nt], false, false);
  }

  const float winv = winv_p[0];

#pragma unroll
  for (int mt = 0; mt < 2; ++mt) {
#pragma unroll
    for (int nt = 0; nt < 4; ++nt) {
      const int n = n0 + nt * 16 + lr;
      const float* lbr = lb + (size_t)n * RDIM;
      float lbv[RDIM];
#pragma unroll
      for (int r = 0; r < RDIM; ++r) lbv[r] = lbr[r];
#pragma unroll
      for (int j = 0; j < 8; ++j) {
        const int m = m0 + mt * 16 + hi * 8 + j;
        const float* tr = t + (size_t)m * RDIM;
        float lora = 0.0f;
#pragma unroll
        for (int r = 0; r < RDIM; ++r) lora = fmaf(tr[r], lbv[r], lora);
        float y = (float)acc[mt][nt][j] * (inv_sx[m] * winv) + lora;
        if (relu) y = fmaxf(y, 0.0f);
        out[(size_t)m * HDIM + n] = y;
      }
    }
  }
}

extern "C" void kernel_launch(void* const* d_in, const int* in_sizes, int n_in,
                              void* d_out, int out_size, void* d_ws,
                              size_t ws_size, hipStream_t stream) {
  (void)in_sizes; (void)n_in; (void)out_size; (void)ws_size;
  const float* x = (const float*)d_in[0];
  uint8_t* ws = (uint8_t*)d_ws;
  size_t off = 0;
  signed char* xq = (signed char*)(ws + off); off += (size_t)MTOK * HDIM;
  signed char* wqb = (signed char*)(ws + off); off += (size_t)HDIM * HDIM;
  float* inv_sx = (float*)(ws + off); off += (size_t)MTOK * sizeof(float);
  float* tbuf = (float*)(ws + off); off += (size_t)MTOK * RDIM * sizeof(float);
  float* wsum = (float*)(ws + off); off += 256;
  float* winv = wsum + 1;

  float* y = (float*)d_out;
  const float* cur = x;
  const int n_w = HDIM * HDIM;

  for (int l = 0; l < 5; ++l) {
    const float* w = (const float*)d_in[1 + 3 * l];
    const float* la = (const float*)d_in[2 + 3 * l];
    const float* lbp = (const float*)d_in[3 + 3 * l];

    zero_scalar_k<<<1, 64, 0, stream>>>(wsum);
    absmean_reduce_k<<<1024, 256, 0, stream>>>(w, wsum, n_w);
    quant_weight_k<<<2048, 256, 0, stream>>>(w, wsum, wqb, winv, n_w,
                                             1.0f / (float)n_w);
    act_quant_lora_k<<<MTOK, 256, 0, stream>>>(cur, la, xq, inv_sx, tbuf);
    bitnet_wmma_gemm_k<<<dim3(HDIM / 128, MTOK / 128), 256, 0, stream>>>(
        xq, inv_sx, wqb, winv, tbuf, lbp, y, (l < 4) ? 1 : 0);
    cur = y;
  }
}
